// GCN_V_46969762349347
// MI455X (gfx1250) — matrix operand order repaired and checked by reading, not compiled, not run on hardware
//
#include <hip/hip_runtime.h>

typedef __attribute__((ext_vector_type(16))) __bf16 v16bf;
typedef __attribute__((ext_vector_type(8)))  float  v8f;

#define DFEAT 256
#define HDIM  512
#define NPAD  10016   // 10000 rounded up to multiple of 32

__device__ __forceinline__ unsigned short f2bf(float f) {
  unsigned int u = __builtin_bit_cast(unsigned int, f);
  u += 0x7FFFu + ((u >> 16) & 1u);      // round-to-nearest-even
  return (unsigned short)(u >> 16);
}
__device__ __forceinline__ float bf2f(unsigned short h) {
  unsigned int u = ((unsigned int)h) << 16;
  return __builtin_bit_cast(float, u);
}

// ---------------- init / convert kernels ----------------
__global__ void zero_f32_kernel(float* __restrict__ p, long n) {
  long i = (long)blockIdx.x * blockDim.x + threadIdx.x;
  if (i < n) p[i] = 0.0f;
}

// convert fp32 -> bf16, zero-filling [n_in, n_total)
__global__ void cvt_pad_kernel(const float* __restrict__ in, unsigned short* __restrict__ out,
                               long n_in, long n_total) {
  long i = (long)blockIdx.x * blockDim.x + threadIdx.x;
  if (i >= n_total) return;
  out[i] = (i < n_in) ? f2bf(in[i]) : (unsigned short)0;
}

// W[K][NC] row-major fp32 -> Wt[NC][K] bf16 (column-major of W)
__global__ void transpose_cvt_kernel(const float* __restrict__ W, unsigned short* __restrict__ Wt,
                                     int K, int NC) {
  long i = (long)blockIdx.x * blockDim.x + threadIdx.x;
  if (i >= (long)K * NC) return;
  int k = (int)(i / NC);
  int n = (int)(i % NC);
  Wt[(size_t)n * K + k] = f2bf(W[i]);
}

// ---------------- SpMM: agg[dst] += val * x[src] ----------------
// one wave32 per edge; D=256 -> 8 floats per lane
__global__ void spmm_kernel(const float* __restrict__ x,
                            const int* __restrict__ esrc, const int* __restrict__ edst,
                            const float* __restrict__ eval, float* __restrict__ agg, int E) {
  int e    = (blockIdx.x * blockDim.x + threadIdx.x) >> 5;
  int lane = threadIdx.x & 31;
  if (e >= E) return;
  int   s = esrc[e];
  int   d = edst[e];
  float v = eval[e];
  const float* xr = x   + (size_t)s * DFEAT;
  float*       ar = agg + (size_t)d * DFEAT;
#pragma unroll
  for (int i = 0; i < DFEAT / 32; ++i) {
    int c = lane + i * 32;
    atomicAdd(&ar[c], v * xr[c]);
  }
}

// ---------------- bf16 WMMA GEMM with fused epilogue ----------------
// One wave computes a 32(M) x 32(N) output tile: 2 A-frags x 2 B-frags -> 4 WMMAs
// per 32-wide K step. K loop over one A source only (no per-iteration select).
__device__ __forceinline__ void accum_panel(const unsigned short* __restrict__ A,
                                            int strideA,
                                            const unsigned short* __restrict__ Bt,
                                            int Kb,       // Bt row stride (total K)
                                            int kbStart,  // starting global k within B
                                            int row0, int col0, int Klen,
                                            int hi16, int l16, v8f acc[2][2]) {
  union Frag { v16bf v; unsigned int u[8]; };
  const unsigned int* ap0 = (const unsigned int*)(A + (size_t)(row0 + l16) * (size_t)strideA);
  const unsigned int* ap1 = (const unsigned int*)(A + (size_t)(row0 + 16 + l16) * (size_t)strideA);
  const unsigned int* bp0 = (const unsigned int*)(Bt + (size_t)(col0 + l16) * (size_t)Kb + kbStart);
  const unsigned int* bp1 = (const unsigned int*)(Bt + (size_t)(col0 + 16 + l16) * (size_t)Kb + kbStart);
#pragma unroll 4
  for (int kk = 0; kk < Klen; kk += 32) {
    Frag a0, a1, b0, b1;
#pragma unroll
    for (int v = 0; v < 8; ++v) {
      // A 16x32 bf16 layout: lanes 0-15: K 0..7 (v0-3), 16..23 (v4-7); lanes 16-31: +8
      int ka = (kk + (((v < 4) ? 0 : 16) + hi16 * 8 + (v & 3) * 2)) >> 1;
      a0.u[v] = ap0[ka];
      a1.u[v] = ap1[ka];
      // B 32x16 bf16 layout: lanes 0-15: K 0..7 (v0-3), 16..23 (v4-7); lanes 16-31: +8
      int kb = (kk + (((v < 4) ? 0 : 16) + hi16 * 8 + (v & 3) * 2)) >> 1;
      b0.u[v] = bp0[kb];
      b1.u[v] = bp1[kb];
    }
    acc[0][0] = __builtin_amdgcn_wmma_f32_16x16x32_bf16(false, a0.v, false, b0.v,
                                                        (short)0, acc[0][0], false, false);
    acc[0][1] = __builtin_amdgcn_wmma_f32_16x16x32_bf16(false, a0.v, false, b1.v,
                                                        (short)0, acc[0][1], false, false);
    acc[1][0] = __builtin_amdgcn_wmma_f32_16x16x32_bf16(false, a1.v, false, b0.v,
                                                        (short)0, acc[1][0], false, false);
    acc[1][1] = __builtin_amdgcn_wmma_f32_16x16x32_bf16(false, a1.v, false, b1.v,
                                                        (short)0, acc[1][1], false, false);
  }
}

// Out[Npad, HO] = act( A[Npad, K] @ B[K, HO] + bias ), act = PReLU(slope); slope=null => ReLU.
// A columns [0, splitK) come from A0, [splitK, K) from A1 (both row stride strideA).
// Bt is B column-major: Bt[n][k], stride K. Block = 128 threads = 4 waves (4 col tiles).
__global__ void gemm_bf16_wmma_kernel(const unsigned short* __restrict__ A0,
                                      const unsigned short* __restrict__ A1,
                                      int strideA, int splitK,
                                      const unsigned short* __restrict__ Bt,
                                      const float* __restrict__ bias,
                                      const float* __restrict__ slope,
                                      unsigned short* __restrict__ Out,
                                      int K, int HO) {
  const int lane = threadIdx.x & 31;
  const int wave = threadIdx.x >> 5;
  const int hi16 = lane >> 4;
  const int l16  = lane & 15;
  const int row0 = blockIdx.x * 32;
  const int col0 = (blockIdx.y * 4 + wave) * 32;

  v8f acc[2][2] = {};

  const int len0 = (splitK < K) ? splitK : K;
  accum_panel(A0, strideA, Bt, K, 0, row0, col0, len0, hi16, l16, acc);
  if (len0 < K)
    accum_panel(A1, strideA, Bt, K, len0, row0, col0, K - len0, hi16, l16, acc);

#pragma unroll
  for (int c = 0; c < 2; ++c) {
    const int   n  = col0 + c * 16 + l16;
    const float bv = bias[n];
    const float sl = slope ? slope[n] : 0.0f;
#pragma unroll
    for (int rr = 0; rr < 2; ++rr) {
#pragma unroll
      for (int r = 0; r < 8; ++r) {
        float val = acc[rr][c][r] + bv;
        val = (val > 0.0f) ? val : sl * val;        // PReLU; slope 0 == ReLU
        int mo = row0 + rr * 16 + hi16 * 8 + r;     // C/D layout: M = r (+8 for hi half)
        Out[(size_t)mo * HO + n] = f2bf(val);
      }
    }
  }
}

// ---------------- final mat-vec: pred = z @ W2 + b2 ----------------
__global__ void matvec_kernel(const unsigned short* __restrict__ z,
                              const float* __restrict__ W2, const float* __restrict__ b2,
                              float* __restrict__ out, int N, int K) {
  int row  = (blockIdx.x * blockDim.x + threadIdx.x) >> 5;
  int lane = threadIdx.x & 31;
  if (row >= N) return;
  const unsigned short* zr = z + (size_t)row * K;
  float sum = 0.0f;
  for (int k = lane; k < K; k += 32) sum += bf2f(zr[k]) * W2[k];
#pragma unroll
  for (int off = 16; off > 0; off >>= 1) sum += __shfl_down(sum, off, 32);
  if (lane == 0) out[row] = sum + b2[0];
}

extern "C" void kernel_launch(void* const* d_in, const int* in_sizes, int n_in,
                              void* d_out, int out_size, void* d_ws, size_t ws_size,
                              hipStream_t stream) {
  const float* x    = (const float*)d_in[0];
  const int*   esrc = (const int*)  d_in[1];
  const int*   edst = (const int*)  d_in[2];
  const float* eval = (const float*)d_in[3];
  const float* Wc   = (const float*)d_in[4];
  const float* bc   = (const float*)d_in[5];
  const float* W1   = (const float*)d_in[6];
  const float* b1   = (const float*)d_in[7];
  const float* pa   = (const float*)d_in[8];
  const float* W2   = (const float*)d_in[9];
  const float* b2   = (const float*)d_in[10];

  const int N = in_sizes[0] / DFEAT;   // 10000
  const int E = in_sizes[1];           // 320000
  float* out = (float*)d_out;

  // workspace layout (256B-aligned sections)
  char* p = (char*)d_ws;
  float*          agg_f32 = (float*)p;          p += (size_t)NPAD * DFEAT * sizeof(float);
  unsigned short* x_bf    = (unsigned short*)p; p += (size_t)NPAD * DFEAT * sizeof(unsigned short);
  unsigned short* agg_bf  = (unsigned short*)p; p += (size_t)NPAD * DFEAT * sizeof(unsigned short);
  unsigned short* Wc_t    = (unsigned short*)p; p += (size_t)2 * DFEAT * HDIM * sizeof(unsigned short);
  unsigned short* W1_t    = (unsigned short*)p; p += (size_t)HDIM * HDIM * sizeof(unsigned short);
  unsigned short* h_bf    = (unsigned short*)p; p += (size_t)NPAD * HDIM * sizeof(unsigned short);
  unsigned short* z_bf    = (unsigned short*)p; p += (size_t)NPAD * HDIM * sizeof(unsigned short);

  const long nAgg = (long)NPAD * DFEAT;
  const long nX   = (long)N * DFEAT;

  zero_f32_kernel<<<(int)((nAgg + 255) / 256), 256, 0, stream>>>(agg_f32, nAgg);
  cvt_pad_kernel<<<(int)((nAgg + 255) / 256), 256, 0, stream>>>(x, x_bf, nX, nAgg);

  spmm_kernel<<<(E * 32 + 255) / 256, 256, 0, stream>>>(x, esrc, edst, eval, agg_f32, E);

  cvt_pad_kernel<<<(int)((nAgg + 255) / 256), 256, 0, stream>>>(agg_f32, agg_bf, nAgg, nAgg);

  const long nWc = (long)2 * DFEAT * HDIM;
  transpose_cvt_kernel<<<(int)((nWc + 255) / 256), 256, 0, stream>>>(Wc, Wc_t, 2 * DFEAT, HDIM);
  const long nW1 = (long)HDIM * HDIM;
  transpose_cvt_kernel<<<(int)((nW1 + 255) / 256), 256, 0, stream>>>(W1, W1_t, HDIM, HDIM);

  dim3 ggrid(NPAD / 32, HDIM / 128);   // 313 row tiles x (16 col tiles / 4 waves)
  // GEMM1: [x | agg] @ W_conv + b_conv, ReLU  (A split at K=256, stride 256)
  gemm_bf16_wmma_kernel<<<ggrid, 128, 0, stream>>>(x_bf, agg_bf, DFEAT, DFEAT,
                                                   Wc_t, bc, nullptr, h_bf,
                                                   2 * DFEAT, HDIM);
  // GEMM2: h @ W1 + b1, PReLU  (no split: splitK > K)
  gemm_bf16_wmma_kernel<<<ggrid, 128, 0, stream>>>(h_bf, h_bf, HDIM, 2 * HDIM,
                                                   W1_t, b1, pa, z_bf,
                                                   HDIM, HDIM);

  matvec_kernel<<<(N * 32 + 255) / 256, 256, 0, stream>>>(z_bf, W2, b2, out, N, HDIM);
}
